// ST_decomposition00_5832565587988
// MI455X (gfx1250) — matrix operand order repaired and checked by reading, not compiled, not run on hardware
//
#include <hip/hip_runtime.h>
#include <hip/hip_bf16.h>
#include <math.h>

typedef __attribute__((ext_vector_type(16))) _Float16 v16h;
typedef __attribute__((ext_vector_type(8)))  _Float16 v8h;
typedef __attribute__((ext_vector_type(8)))  float    v8f;

// ---- problem constants (match reference) ----
constexpr int cB = 4, cL = 96, cN = 128, cZ = 64, cK = 16, cFF = 256;
constexpr int cR = cB * cL * cN;     // 49152 rows (== B*N*L == B*L*N)
constexpr int cD = cZ * cN;          // 8192 VQ features
constexpr int cT = cB * cL;          // 384 pattern length

// ---- output layout (floats) ----
constexpr size_t OFF_W  = 0;                       // weighted (B,L,N,Z)
constexpr size_t OFF_CB = (size_t)cR * cZ;         // codebook2 (K, T)
constexpr size_t OFF_D  = OFF_CB + (size_t)cK*cT;  // dists (D, K)
constexpr size_t OFF_SC = OFF_D + (size_t)cD*cK;   // 5 scalars

// ---- workspace layout (float units) ----
constexpr size_t W_A    = 0;
constexpr size_t W_S    = W_A    + (size_t)cR*cZ;
constexpr size_t W_X    = W_S    + (size_t)cR*cZ;
constexpr size_t W_Q    = W_X    + (size_t)cR*cZ;
constexpr size_t W_K    = W_Q    + (size_t)cR*cZ;
constexpr size_t W_V    = W_K    + (size_t)cR*cZ;
constexpr size_t W_T    = W_V    + (size_t)cR*cZ;
constexpr size_t W_G    = W_T    + (size_t)cR*cZ;
constexpr size_t W_F    = W_G    + (size_t)cR*cZ;
constexpr size_t W_SOFT = W_F    + (size_t)cR*cFF;
constexpr size_t W_ACC  = W_SOFT + (size_t)cR*cK;        // 8 float accumulators
constexpr size_t W_INT  = W_ACC  + 8;                    // bins1(16) bins2(16) idx1(D) idx2(D)
constexpr size_t W_HACT = W_INT  + 32 + 2*(size_t)cD;    // f16 activation mirror (R*FF halves)
constexpr size_t W_HWT  = W_HACT + (size_t)cR*cFF/2;     // f16 transposed weights (FF*Z halves)

// =====================================================================
// f32 -> f16 cast (activations), 4 elems/thread, vector load
// =====================================================================
__global__ void castf2h_kernel(const float* __restrict__ src, _Float16* __restrict__ dst, size_t n)
{
    size_t i = ((size_t)blockIdx.x * 256 + threadIdx.x) * 4;
    if (i + 3 < n) {
        float4 v = *(const float4*)(src + i);
        dst[i+0] = (_Float16)v.x; dst[i+1] = (_Float16)v.y;
        dst[i+2] = (_Float16)v.z; dst[i+3] = (_Float16)v.w;
    }
}

// W (K,N) f32 -> W^T (N,K) f16
__global__ void castWt_kernel(const float* __restrict__ src, _Float16* __restrict__ dst,
                              int Kd, int Nc)
{
    int id = blockIdx.x * 256 + threadIdx.x;
    if (id >= Kd * Nc) return;
    int k = id / Nc, n = id % Nc;
    dst[(size_t)n * Kd + k] = (_Float16)src[id];
}

// =====================================================================
// WMMA GEMM: C[M,N] = A[M,K] * W[K,N] (+bias)(+resid)(relu)
// wave computes a 16 x (16*NT) tile; A fragment reused across NT wmmas.
// Ah: f16 row-major (M,K); Wth: f16 row-major W^T (N,K).
// =====================================================================
template<int NT, bool HASB, bool RESID, bool RELU>
__global__ __launch_bounds__(32)
void wmma_gemm_kernel(const _Float16* __restrict__ Ah, const _Float16* __restrict__ Wth,
                      const float* __restrict__ bias, const float* __restrict__ resid,
                      float* __restrict__ C, int M, int Nc, int Kd)
{
    const int lane  = threadIdx.x;
    const int half  = lane >> 4;
    const int mrow  = blockIdx.x * 16 + (lane & 15);
    const int ncol0 = blockIdx.y * (16 * NT) + (lane & 15);
    v8f acc[NT];
#pragma unroll
    for (int t = 0; t < NT; ++t) acc[t] = (v8f){};
    const _Float16* arow = Ah + (size_t)mrow * Kd;
    for (int k0 = 0; k0 < Kd; k0 += 32) {
        // A (16x32 f16): lanes0-15 K=0..15, lanes16-31 K=16..31
        v8h a0 = *(const v8h*)(arow + k0 + half * 16);
        v8h a1 = *(const v8h*)(arow + k0 + half * 16 + 8);
        v16h a;
#pragma unroll
        for (int i = 0; i < 8; ++i) { a[i] = a0[i]; a[8 + i] = a1[i]; }
#pragma unroll
        for (int t = 0; t < NT; ++t) {
            // B (32x16 f16): lanes0-15 rows K=0..15, lanes16-31 rows K=16..31
            const _Float16* brow = Wth + (size_t)(ncol0 + 16 * t) * Kd + k0 + half * 16;
            v8h b0 = *(const v8h*)(brow);
            v8h b1 = *(const v8h*)(brow + 8);
            v16h b;
#pragma unroll
            for (int i = 0; i < 8; ++i) { b[i] = b0[i]; b[8 + i] = b1[i]; }
            acc[t] = __builtin_amdgcn_wmma_f32_16x16x32_f16(false, a, false, b, (short)0,
                                                            acc[t], false, false);
        }
    }
#pragma unroll
    for (int t = 0; t < NT; ++t) {
        int ncol = ncol0 + 16 * t;
        float bv = HASB ? bias[ncol] : 0.f;
#pragma unroll
        for (int r = 0; r < 8; ++r) {
            int row = blockIdx.x * 16 + half * 8 + r;    // C: VGPR r holds M = half*8+r
            size_t idx = (size_t)row * Nc + ncol;
            float v = acc[t][r] + bv;
            if (RESID) v += resid[idx];
            if (RELU)  v = fmaxf(v, 0.f);
            C[idx] = v;
        }
    }
}

// ============================ small kernels ============================
__global__ void zero_kernel(float* acc, int* ib)
{
    int t = threadIdx.x;
    if (t < 8)  acc[t] = 0.f;
    if (t < 32) ib[t]  = 0;
}

__global__ void inproj_kernel(const float* __restrict__ x, const float* __restrict__ w_in,
                              const float* __restrict__ b_in, float* __restrict__ outA)
{
    size_t id = (size_t)blockIdx.x * 256 + threadIdx.x;
    if (id >= (size_t)cR * cZ) return;
    int z = id % cZ; size_t row = id / cZ;       // row = (b*N+n)*L+l
    int l = row % cL; size_t bn = row / cL;
    int n = bn % cN; int b = bn / cN;
    float xv = x[((size_t)b * cL + l) * cN + n];
    outA[id] = w_in[z] * xv + b_in[z];
}

__global__ void transpose_kernel(const float* __restrict__ in, float* __restrict__ out)
{
    // (B,N,L,Z) rows -> (B,L,N,Z) rows
    size_t id = (size_t)blockIdx.x * 256 + threadIdx.x;
    if (id >= (size_t)cR * cZ) return;
    int z = id % cZ; size_t ro = id / cZ;        // out row = (b*L+l)*N+n
    int n = ro % cN; size_t j = ro / cN;
    int b = j / cL, l = j % cL;
    out[id] = in[(((size_t)b * cN + n) * cL + l) * cZ + z];
}

__global__ void ln_rows_kernel(float* __restrict__ x, const float* __restrict__ g,
                               const float* __restrict__ b, int M)
{
    int r = blockIdx.x * blockDim.x + threadIdx.x;
    if (r >= M) return;
    float* p = x + (size_t)r * cZ;
    float m = 0.f;
    for (int i = 0; i < cZ; ++i) m += p[i];
    m *= (1.f / cZ);
    float v = 0.f;
    for (int i = 0; i < cZ; ++i) { float d = p[i] - m; v += d * d; }
    v *= (1.f / cZ);
    float inv = rsqrtf(v + 1e-5f);
    for (int i = 0; i < cZ; ++i) p[i] = (p[i] - m) * inv * g[i] + b[i];
}

// per-(sequence, head) attention, dh = 16, optional adjacency mask
template<int TT, bool MASK>
__global__ __launch_bounds__(128)
void attn_kernel(const float* __restrict__ Q, const float* __restrict__ Km,
                 const float* __restrict__ V, float* __restrict__ O,
                 const float* __restrict__ adj)
{
    __shared__ float ks[TT * 16];
    __shared__ float vs[TT * 16];
    int s = blockIdx.x, h = blockIdx.y;
    size_t base = (size_t)s * TT * cZ + h * 16;
    for (int i = threadIdx.x; i < TT * 16; i += blockDim.x) {
        int row = i >> 4, d = i & 15;
        ks[i] = Km[base + (size_t)row * cZ + d];
        vs[i] = V [base + (size_t)row * cZ + d];
    }
    __syncthreads();
    int t = threadIdx.x;
    if (t >= TT) return;
    float q[16];
#pragma unroll
    for (int d = 0; d < 16; ++d) q[d] = Q[base + (size_t)t * cZ + d];
    float mx = -1e30f;
    for (int kr = 0; kr < TT; ++kr) {
        float sc = 0.f;
#pragma unroll
        for (int d = 0; d < 16; ++d) sc += q[d] * ks[kr * 16 + d];
        sc *= 0.25f;
        if (MASK) sc = (adj[t * cN + kr] > 0.f) ? sc : -1e9f;
        mx = fmaxf(mx, sc);
    }
    float ssum = 0.f;
    for (int kr = 0; kr < TT; ++kr) {
        float sc = 0.f;
#pragma unroll
        for (int d = 0; d < 16; ++d) sc += q[d] * ks[kr * 16 + d];
        sc *= 0.25f;
        if (MASK) sc = (adj[t * cN + kr] > 0.f) ? sc : -1e9f;
        ssum += __expf(sc - mx);
    }
    float inv = 1.f / ssum;
    float o[16] = {};
    for (int kr = 0; kr < TT; ++kr) {
        float sc = 0.f;
#pragma unroll
        for (int d = 0; d < 16; ++d) sc += q[d] * ks[kr * 16 + d];
        sc *= 0.25f;
        if (MASK) sc = (adj[t * cN + kr] > 0.f) ? sc : -1e9f;
        float w = __expf(sc - mx) * inv;
#pragma unroll
        for (int d = 0; d < 16; ++d) o[d] += w * vs[kr * 16 + d];
    }
#pragma unroll
    for (int d = 0; d < 16; ++d) O[base + (size_t)t * cZ + d] = o[d];
}

// spatial attention over nodes, full Z, residual
__global__ __launch_bounds__(128)
void spatial_attn_kernel(const float* __restrict__ Qb, const float* __restrict__ Kb,
                         const float* __restrict__ Vb, const float* __restrict__ Xin,
                         float* __restrict__ Xout)
{
    __shared__ float ksh[cN * cZ];
    __shared__ float vsh[cN * cZ];
    int s = blockIdx.x, n = threadIdx.x;
    size_t base = (size_t)s * cN * cZ;
    for (int i = n; i < cN * cZ; i += 128) { ksh[i] = Kb[base + i]; vsh[i] = Vb[base + i]; }
    __syncthreads();
    float q[cZ];
    for (int z = 0; z < cZ; ++z) q[z] = Qb[base + (size_t)n * cZ + z];
    float sc[cN]; float mx = -1e30f;
    for (int m = 0; m < cN; ++m) {
        float d = 0.f;
        for (int z = 0; z < cZ; ++z) d += q[z] * ksh[m * cZ + z];
        d *= 0.125f;   // 1/sqrt(64)
        sc[m] = d; mx = fmaxf(mx, d);
    }
    float ssum = 0.f;
    for (int m = 0; m < cN; ++m) { float w = __expf(sc[m] - mx); sc[m] = w; ssum += w; }
    float inv = 1.f / ssum;
    for (int z = 0; z < cZ; ++z) {
        float o = 0.f;
        for (int m = 0; m < cN; ++m) o += sc[m] * vsh[m * cZ + z];
        Xout[base + (size_t)n * cZ + z] = Xin[base + (size_t)n * cZ + z] + o * inv;
    }
}

// GAT layer: e = lrelu(e1[n]+e2[m]); adj mask; softmax; elu(att@Wh)
__global__ __launch_bounds__(128)
void gat_layer_kernel(const float* __restrict__ Wh, const float* __restrict__ a,
                      const float* __restrict__ adj, float* __restrict__ out)
{
    __shared__ float wsh[cN * cZ];
    __shared__ float e2s[cN];
    int s = blockIdx.x, n = threadIdx.x;
    size_t base = (size_t)s * cN * cZ;
    for (int i = n; i < cN * cZ; i += 128) wsh[i] = Wh[base + i];
    __syncthreads();
    float e1 = 0.f, e2 = 0.f;
    for (int z = 0; z < cZ; ++z) { float w = wsh[n * cZ + z]; e1 += w * a[z]; e2 += w * a[cZ + z]; }
    e2s[n] = e2;
    __syncthreads();
    float att[cN]; float mx = -1e30f;
    for (int m = 0; m < cN; ++m) {
        float e = e1 + e2s[m];
        e = (e > 0.f) ? e : 0.2f * e;                 // leaky relu
        e = (adj[n * cN + m] > 0.f) ? e : -9e15f;
        att[m] = e; mx = fmaxf(mx, e);
    }
    float ssum = 0.f;
    for (int m = 0; m < cN; ++m) { float w = __expf(att[m] - mx); att[m] = w; ssum += w; }
    float inv = 1.f / ssum;
    for (int z = 0; z < cZ; ++z) {
        float o = 0.f;
        for (int m = 0; m < cN; ++m) o += att[m] * wsh[m * cZ + z];
        o *= inv;
        out[base + (size_t)n * cZ + z] = (o > 0.f) ? o : (__expf(o) - 1.f);  // elu
    }
}

__global__ void softmax16_kernel(float* __restrict__ soft, int M)
{
    int r = blockIdx.x * blockDim.x + threadIdx.x;
    if (r >= M) return;
    float* p = soft + (size_t)r * cK;
    float mx = -1e30f;
    for (int k = 0; k < cK; ++k) mx = fmaxf(mx, p[k]);
    float s = 0.f;
    for (int k = 0; k < cK; ++k) { float w = __expf(p[k] - mx); p[k] = w; s += w; }
    float inv = 1.f / s;
    for (int k = 0; k < cK; ++k) p[k] *= inv;
}

// VQ assignment: argmin_k ||flat_d - emb_k||  (||flat||^2 dropped - constant per d)
__global__ void vq_assign_kernel(const float* __restrict__ X, const float* __restrict__ emb_in,
                                 const int* __restrict__ idx_in, const float* __restrict__ emb,
                                 int* __restrict__ idx_out, int* __restrict__ bins, int mode)
{
    int d = blockIdx.x * blockDim.x + threadIdx.x;
    if (d >= cD) return;
    int zz = d / cN, nn = d % cN;
    float best = 3.4e38f; int bi = 0;
    for (int k = 0; k < cK; ++k) {
        const float* ek = emb + (size_t)k * cT;
        float dot = 0.f, ee = 0.f;
        if (mode == 0) {
            for (int j = 0; j < cT; ++j) {
                float e = ek[j];
                float f = X[((size_t)j * cN + nn) * cZ + zz];
                dot += e * f; ee += e * e;
            }
        } else {
            const float* f1 = emb_in + (size_t)idx_in[d] * cT;
            for (int j = 0; j < cT; ++j) { float e = ek[j]; dot += e * f1[j]; ee += e * e; }
        }
        float dd = ee - 2.f * dot;
        if (dd < best) { best = dd; bi = k; }
    }
    idx_out[d] = bi;
    atomicAdd(&bins[bi], 1);
}

__global__ __launch_bounds__(256)
void vq_loss_kernel(const float* __restrict__ X, const float* __restrict__ emb1,
                    const float* __restrict__ emb2, const int* __restrict__ idx1,
                    const int* __restrict__ idx2, float* __restrict__ acc)
{
    __shared__ float red[256];
    size_t id = (size_t)blockIdx.x * 256 + threadIdx.x;   // over T*D
    float s1 = 0.f, s2 = 0.f;
    if (id < (size_t)cT * cD) {
        int j = id / cD, d = id % cD;
        int zz = d / cN, nn = d % cN;
        float in = X[((size_t)j * cN + nn) * cZ + zz];
        float q1 = emb1[(size_t)idx1[d] * cT + j];
        float q2 = emb2[(size_t)idx2[d] * cT + j];
        float d1 = q1 - in, d2v = q2 - q1;
        s1 = d1 * d1; s2 = d2v * d2v;
    }
    red[threadIdx.x] = s1; __syncthreads();
    for (int st = 128; st > 0; st >>= 1) { if (threadIdx.x < st) red[threadIdx.x] += red[threadIdx.x + st]; __syncthreads(); }
    if (threadIdx.x == 0) atomicAdd(&acc[2], red[0]);
    __syncthreads();
    red[threadIdx.x] = s2; __syncthreads();
    for (int st = 128; st > 0; st >>= 1) { if (threadIdx.x < st) red[threadIdx.x] += red[threadIdx.x + st]; __syncthreads(); }
    if (threadIdx.x == 0) atomicAdd(&acc[3], red[0]);
}

// weighted = z4 - (soft . centers); also accumulate sum|weighted|
__global__ __launch_bounds__(256)
void weighted_kernel(const float* __restrict__ X, const float* __restrict__ soft,
                     const float* __restrict__ cb2, float* __restrict__ wout,
                     float* __restrict__ acc)
{
    __shared__ float red[256];
    int id = blockIdx.x * 256 + threadIdx.x;   // row (b*L+l)*N+n
    float asum = 0.f;
    if (id < cR) {
        int j = id / cN;
        float c = 0.f;
        const float* sp = soft + (size_t)id * cK;
        for (int k = 0; k < cK; ++k) c += sp[k] * cb2[(size_t)k * cT + j];
        const float* xp = X + (size_t)id * cZ;
        float* wp = wout + (size_t)id * cZ;
        for (int z = 0; z < cZ; ++z) { float w = xp[z] - c; wp[z] = w; asum += fabsf(w); }
    }
    red[threadIdx.x] = asum; __syncthreads();
    for (int st = 128; st > 0; st >>= 1) { if (threadIdx.x < st) red[threadIdx.x] += red[threadIdx.x + st]; __syncthreads(); }
    if (threadIdx.x == 0) atomicAdd(&acc[1], red[0]);
}

// dec1: 1x1 conv on quantized (gathered from both codebooks) + BN + relu
__global__ __launch_bounds__(64)
void dec1_kernel(const float* __restrict__ emb1, const float* __restrict__ emb2,
                 const int* __restrict__ idx1, const int* __restrict__ idx2,
                 const float* __restrict__ w, const float* __restrict__ bvec,
                 const float* __restrict__ bng, const float* __restrict__ bnb,
                 float* __restrict__ h1)
{
    __shared__ float qv[cZ];
    int blk = blockIdx.x;                      // (b*N+n)*L + l
    int l = blk % cL; int bn = blk / cL;
    int n = bn % cN;  int b = bn / cN;
    int j = b * cL + l;
    int t = threadIdx.x;                       // channel index
    int d = t * cN + n;
    qv[t] = emb1[(size_t)idx1[d] * cT + j] + emb2[(size_t)idx2[d] * cT + j];
    __syncthreads();
    float acc = 0.f;
    for (int c = 0; c < cZ; ++c) acc += w[t * cZ + c] * qv[c];
    acc += bvec[t];
    acc = acc * rsqrtf(1.00001f) * bng[t] + bnb[t];
    acc = acc > 0.f ? acc : 0.f;
    h1[(((size_t)b * cZ + t) * cN + n) * cL + l] = acc;
}

// 3x3 SAME conv over (N,L) + bias + BN + relu
__global__ __launch_bounds__(256)
void conv2_kernel(const float* __restrict__ h1, const float* __restrict__ w,
                  const float* __restrict__ bias, const float* __restrict__ bng,
                  const float* __restrict__ bnb, float* __restrict__ out)
{
    size_t id = (size_t)blockIdx.x * 256 + threadIdx.x;
    if (id >= (size_t)cB * cZ * cN * cL) return;
    int l = id % cL; size_t r = id / cL;
    int n = r % cN; r /= cN;
    int o = r % cZ; int b = r / cZ;
    float acc = 0.f;
    for (int c = 0; c < cZ; ++c) {
        const float* hp = h1 + ((size_t)b * cZ + c) * cN * cL;
        const float* wp = w + ((size_t)o * cZ + c) * 9;
        for (int dn = -1; dn <= 1; ++dn) {
            int nn = n + dn; if (nn < 0 || nn >= cN) continue;
            for (int dl = -1; dl <= 1; ++dl) {
                int ll = l + dl; if (ll < 0 || ll >= cL) continue;
                acc += wp[(dn + 1) * 3 + (dl + 1)] * hp[(size_t)nn * cL + ll];
            }
        }
    }
    acc += bias[o];
    acc = acc * rsqrtf(1.00001f) * bng[o] + bnb[o];
    out[id] = acc > 0.f ? acc : 0.f;
}

// residual decoder (recomputed inline) + combine + output projection + recon SSE
__global__ __launch_bounds__(256)
void final_kernel(const float* __restrict__ dq, const float* __restrict__ weighted,
                  const float* __restrict__ resw, const float* __restrict__ resb,
                  const float* __restrict__ bng, const float* __restrict__ bnb,
                  const float* __restrict__ outw, const float* __restrict__ outb,
                  const float* __restrict__ x, float* __restrict__ acc)
{
    __shared__ float rw[cZ * cZ];
    __shared__ float red[256];
    for (int i = threadIdx.x; i < cZ * cZ; i += 256) rw[i] = resw[i];
    __syncthreads();
    int id = blockIdx.x * 256 + threadIdx.x;   // (b*N+n)*L + l
    float se = 0.f;
    if (id < cR) {
        int l = id % cL; int r = id / cL;
        int n = r % cN; int b = r / cN;
        int j = b * cL + l;
        float wv[cZ];
        const float* wp = weighted + ((size_t)j * cN + n) * cZ;
        for (int c = 0; c < cZ; ++c) wv[c] = wp[c];
        const float s = rsqrtf(1.00001f);
        float accum = 0.f;
        for (int o2 = 0; o2 < cZ; ++o2) {
            float dr = 0.f;
            for (int c = 0; c < cZ; ++c) dr += rw[o2 * cZ + c] * wv[c];
            dr += resb[o2];
            dr = dr * s * bng[o2] + bnb[o2];
            dr = dr > 0.f ? dr : 0.f;
            float dqv = dq[(((size_t)b * cZ + o2) * cN + n) * cL + l];
            accum += outw[o2] * (dqv + dr);
        }
        accum += outb[0];
        float diff = accum - x[((size_t)b * cL + l) * cN + n];
        se = diff * diff;
    }
    red[threadIdx.x] = se; __syncthreads();
    for (int st = 128; st > 0; st >>= 1) { if (threadIdx.x < st) red[threadIdx.x] += red[threadIdx.x + st]; __syncthreads(); }
    if (threadIdx.x == 0) atomicAdd(&acc[0], red[0]);
}

__global__ void dists_kernel(const float* __restrict__ S, const float* __restrict__ cb2,
                             float* __restrict__ dout)
{
    int i = blockIdx.x * blockDim.x + threadIdx.x;
    if (i >= cD) return;
    int zz = i / cN, nn = i % cN;
    float ff = 0.f;
    for (int j = 0; j < cT; ++j) { float f = S[((size_t)j * cN + nn) * cZ + zz]; ff += f * f; }
    for (int k = 0; k < cK; ++k) {
        const float* ek = cb2 + (size_t)k * cT;
        float dot = 0.f, ee = 0.f;
        for (int j = 0; j < cT; ++j) {
            float e = ek[j];
            float f = S[((size_t)j * cN + nn) * cZ + zz];
            dot += e * f; ee += e * e;
        }
        float d2 = ff - 2.f * dot + ee;
        dout[(size_t)i * cK + k] = sqrtf(fmaxf(d2, 0.f));
    }
}

__global__ void copycb_kernel(const float* __restrict__ src, float* __restrict__ dst)
{
    int i = blockIdx.x * 256 + threadIdx.x;
    if (i < cK * cT) dst[i] = src[i];
}

__global__ void scalars_kernel(const float* __restrict__ acc, const int* __restrict__ bins1,
                               const int* __restrict__ bins2, float* __restrict__ sout)
{
    if (threadIdx.x != 0 || blockIdx.x != 0) return;
    float recon = acc[0] / (float)cR;
    float quant = 1.25f * (acc[2] + acc[3]) / (float)((size_t)cT * cD);
    float spars = acc[1] / (float)((size_t)cR * cZ);
    float h1 = 0.f, h2 = 0.f;
    for (int k = 0; k < cK; ++k) {
        float p1 = bins1[k] / (float)cD;
        float p2 = bins2[k] / (float)cD;
        h1 -= p1 * __logf(p1 + 1e-10f);
        h2 -= p2 * __logf(p2 + 1e-10f);
    }
    float ploss = -0.1f * (__logf(__expf(h1) + 1e-10f) + __logf(__expf(h2) + 1e-10f));
    float total = recon + quant + 0.1f * spars + ploss;
    sout[0] = total; sout[1] = recon; sout[2] = quant; sout[3] = spars; sout[4] = ploss;
}

// =====================================================================
extern "C" void kernel_launch(void* const* d_in, const int* in_sizes, int n_in,
                              void* d_out, int out_size, void* d_ws, size_t ws_size,
                              hipStream_t stream)
{
    (void)in_sizes; (void)n_in; (void)out_size; (void)ws_size;
    const float* x        = (const float*)d_in[0];
    const float* adj      = (const float*)d_in[1];
    const float* w_in     = (const float*)d_in[2];
    const float* b_in     = (const float*)d_in[3];
    const float* t_attn_w = (const float*)d_in[4];
    const float* t_attn_b = (const float*)d_in[5];
    const float* t_ffn_w1 = (const float*)d_in[6];
    const float* t_ffn_b1 = (const float*)d_in[7];
    const float* t_ffn_w2 = (const float*)d_in[8];
    const float* t_ffn_b2 = (const float*)d_in[9];
    const float* t_ln     = (const float*)d_in[10];
    const float* f_attn_w = (const float*)d_in[11];
    const float* f_attn_b = (const float*)d_in[12];
    const float* f_ffn_w1 = (const float*)d_in[13];
    const float* f_ffn_b1 = (const float*)d_in[14];
    const float* f_ffn_w2 = (const float*)d_in[15];
    const float* f_ffn_b2 = (const float*)d_in[16];
    const float* f_ln     = (const float*)d_in[17];
    const float* sa_w     = (const float*)d_in[18];
    const float* sa_b     = (const float*)d_in[19];
    const float* gat_W    = (const float*)d_in[20];
    const float* gat_a    = (const float*)d_in[21];
    const float* vq1_emb  = (const float*)d_in[22];
    const float* vq2_emb  = (const float*)d_in[23];
    const float* sc_w1    = (const float*)d_in[24];
    const float* sc_b1    = (const float*)d_in[25];
    const float* sc_w2    = (const float*)d_in[26];
    const float* sc_b2    = (const float*)d_in[27];
    const float* dec1_w   = (const float*)d_in[28];
    const float* dec1_b   = (const float*)d_in[29];
    const float* dec2_w   = (const float*)d_in[30];
    const float* dec2_b   = (const float*)d_in[31];
    const float* res_w    = (const float*)d_in[32];
    const float* res_b    = (const float*)d_in[33];
    const float* bn_g     = (const float*)d_in[34];
    const float* bn_b     = (const float*)d_in[35];
    const float* out_w    = (const float*)d_in[36];
    const float* out_b    = (const float*)d_in[37];

    float* out = (float*)d_out;
    float* ws  = (float*)d_ws;
    float* bufA = ws + W_A;  float* bufS = ws + W_S;  float* bufX = ws + W_X;
    float* bufQ = ws + W_Q;  float* bufK = ws + W_K;  float* bufV = ws + W_V;
    float* bufT = ws + W_T;  float* bufG = ws + W_G;  float* bufF = ws + W_F;
    float* bufSoft = ws + W_SOFT;
    float* acc = ws + W_ACC;
    int* ib    = (int*)(ws + W_INT);
    int* bins1 = ib;         int* bins2 = ib + 16;
    int* idx1  = ib + 32;    int* idx2  = ib + 32 + cD;
    _Float16* hAct = (_Float16*)(ws + W_HACT);
    _Float16* hWt  = (_Float16*)(ws + W_HWT);

    auto gemm = [&](const float* A, const float* W, const float* bias, const float* resid,
                    float* C, int M, int Nc, int Kd, int relu) {
        size_t nA = (size_t)M * Kd;
        castf2h_kernel<<<(int)((nA / 4 + 255) / 256), 256, 0, stream>>>(A, hAct, nA);
        castWt_kernel<<<(Kd * Nc + 255) / 256, 256, 0, stream>>>(W, hWt, Kd, Nc);
        if (Nc % 64 == 0) {
            dim3 g(M / 16, Nc / 64);
            if (!bias)
                wmma_gemm_kernel<4, false, false, false><<<g, 32, 0, stream>>>(hAct, hWt, bias, resid, C, M, Nc, Kd);
            else if (resid)
                wmma_gemm_kernel<4, true, true, false><<<g, 32, 0, stream>>>(hAct, hWt, bias, resid, C, M, Nc, Kd);
            else if (relu)
                wmma_gemm_kernel<4, true, false, true><<<g, 32, 0, stream>>>(hAct, hWt, bias, resid, C, M, Nc, Kd);
            else
                wmma_gemm_kernel<4, true, false, false><<<g, 32, 0, stream>>>(hAct, hWt, bias, resid, C, M, Nc, Kd);
        } else {
            dim3 g(M / 16, Nc / 16);
            wmma_gemm_kernel<1, true, false, false><<<g, 32, 0, stream>>>(hAct, hWt, bias, resid, C, M, Nc, Kd);
        }
    };
    const int elemBlocks = (int)(((size_t)cR * cZ + 255) / 256);
    const int rowBlocks  = (cR + 255) / 256;

    zero_kernel<<<1, 64, 0, stream>>>(acc, ib);

    // --- input projection -> ht (B,N,L,Z) rows ---
    inproj_kernel<<<elemBlocks, 256, 0, stream>>>(x, w_in, b_in, bufA);

    // --- temporal transformer encoder layer ---
    gemm(bufA, t_attn_w + 0 * cZ * cZ, t_attn_b + 0 * cZ, nullptr, bufQ, cR, cZ, cZ, 0);
    gemm(bufA, t_attn_w + 1 * cZ * cZ, t_attn_b + 1 * cZ, nullptr, bufK, cR, cZ, cZ, 0);
    gemm(bufA, t_attn_w + 2 * cZ * cZ, t_attn_b + 2 * cZ, nullptr, bufV, cR, cZ, cZ, 0);
    attn_kernel<cL, false><<<dim3(cB * cN, 4), 128, 0, stream>>>(bufQ, bufK, bufV, bufT, nullptr);
    gemm(bufT, t_attn_w + 3 * cZ * cZ, t_attn_b + 3 * cZ, bufA, bufA, cR, cZ, cZ, 0);
    ln_rows_kernel<<<rowBlocks, 256, 0, stream>>>(bufA, t_ln + 0, t_ln + cZ, cR);
    gemm(bufA, t_ffn_w1, t_ffn_b1, nullptr, bufF, cR, cFF, cZ, 1);
    gemm(bufF, t_ffn_w2, t_ffn_b2, bufA, bufA, cR, cZ, cFF, 0);
    ln_rows_kernel<<<rowBlocks, 256, 0, stream>>>(bufA, t_ln + 2 * cZ, t_ln + 3 * cZ, cR);

    // --- relayout (B,N,L,Z) -> (B,L,N,Z) ---
    transpose_kernel<<<elemBlocks, 256, 0, stream>>>(bufA, bufS);

    // --- feature transformer encoder layer (adjacency-masked) ---
    gemm(bufS, f_attn_w + 0 * cZ * cZ, f_attn_b + 0 * cZ, nullptr, bufQ, cR, cZ, cZ, 0);
    gemm(bufS, f_attn_w + 1 * cZ * cZ, f_attn_b + 1 * cZ, nullptr, bufK, cR, cZ, cZ, 0);
    gemm(bufS, f_attn_w + 2 * cZ * cZ, f_attn_b + 2 * cZ, nullptr, bufV, cR, cZ, cZ, 0);
    attn_kernel<cN, true><<<dim3(cB * cL, 4), 128, 0, stream>>>(bufQ, bufK, bufV, bufT, adj);
    gemm(bufT, f_attn_w + 3 * cZ * cZ, f_attn_b + 3 * cZ, bufS, bufS, cR, cZ, cZ, 0);
    ln_rows_kernel<<<rowBlocks, 256, 0, stream>>>(bufS, f_ln + 0, f_ln + cZ, cR);
    gemm(bufS, f_ffn_w1, f_ffn_b1, nullptr, bufF, cR, cFF, cZ, 1);
    gemm(bufF, f_ffn_w2, f_ffn_b2, bufS, bufS, cR, cZ, cFF, 0);
    ln_rows_kernel<<<rowBlocks, 256, 0, stream>>>(bufS, f_ln + 2 * cZ, f_ln + 3 * cZ, cR);
    // bufS now == s_feat (kept for dists)

    // --- spatial attention (residual) -> bufX ---
    gemm(bufS, sa_w + 0 * cZ * cZ, sa_b + 0 * cZ, nullptr, bufQ, cR, cZ, cZ, 0);
    gemm(bufS, sa_w + 1 * cZ * cZ, sa_b + 1 * cZ, nullptr, bufK, cR, cZ, cZ, 0);
    gemm(bufS, sa_w + 2 * cZ * cZ, sa_b + 2 * cZ, nullptr, bufV, cR, cZ, cZ, 0);
    spatial_attn_kernel<<<cB * cL, 128, 0, stream>>>(bufQ, bufK, bufV, bufS, bufX);

    // --- dual VQ ---
    vq_assign_kernel<<<cD / 256, 256, 0, stream>>>(bufX, nullptr, nullptr, vq1_emb, idx1, bins1, 0);
    vq_assign_kernel<<<cD / 256, 256, 0, stream>>>(bufX, vq1_emb, idx1, vq2_emb, idx2, bins2, 1);
    vq_loss_kernel<<<(int)(((size_t)cT * cD + 255) / 256), 256, 0, stream>>>(bufX, vq1_emb, vq2_emb, idx1, idx2, acc);

    // --- 2 GAT layers ---
    gemm(bufX, gat_W + 0 * cZ * cZ, nullptr, nullptr, bufT, cR, cZ, cZ, 0);
    gat_layer_kernel<<<cB * cL, 128, 0, stream>>>(bufT, gat_a + 0, adj, bufG);
    gemm(bufG, gat_W + 1 * cZ * cZ, nullptr, nullptr, bufT, cR, cZ, cZ, 0);
    gat_layer_kernel<<<cB * cL, 128, 0, stream>>>(bufT, gat_a + 2 * cZ, adj, bufG);

    // --- score head: soft = softmax(relu(h@w1+b1)@w2+b2) ---
    gemm(bufG, sc_w1, sc_b1, nullptr, bufA, cR, cZ, cZ, 1);
    gemm(bufA, sc_w2, sc_b2, nullptr, bufSoft, cR, cK, cZ, 0);
    softmax16_kernel<<<rowBlocks, 256, 0, stream>>>(bufSoft, cR);

    // --- weighted residual (output #1) + sparsity ---
    weighted_kernel<<<rowBlocks, 256, 0, stream>>>(bufX, bufSoft, vq2_emb, out + OFF_W, acc);

    // --- conv decoder on quantized ---
    dec1_kernel<<<cR, 64, 0, stream>>>(vq1_emb, vq2_emb, idx1, idx2, dec1_w, dec1_b,
                                       bn_g + 0, bn_b + 0, bufQ);
    conv2_kernel<<<(int)(((size_t)cB * cZ * cN * cL + 255) / 256), 256, 0, stream>>>(
        bufQ, dec2_w, dec2_b, bn_g + cZ, bn_b + cZ, bufK);

    // --- residual decoder + combine + output proj + recon ---
    final_kernel<<<rowBlocks, 256, 0, stream>>>(bufK, out + OFF_W, res_w, res_b,
                                                bn_g + 2 * cZ, bn_b + 2 * cZ,
                                                out_w, out_b, x, acc);

    // --- codebook2 copy (output #2) + dists (output #3) + scalars ---
    copycb_kernel<<<(cK * cT + 255) / 256, 256, 0, stream>>>(vq2_emb, out + OFF_CB);
    dists_kernel<<<cD / 256, 256, 0, stream>>>(bufS, vq2_emb, out + OFF_D);
    scalars_kernel<<<1, 1, 0, stream>>>(acc, bins1, bins2, out + OFF_SC);
}
